// GNNReachabilityNet_25082609009421
// MI455X (gfx1250) — matrix operand order repaired and checked by reading, not compiled, not run on hardware
//
#include <hip/hip_runtime.h>
#include <cstdint>
#include <cstddef>

#define BB 8
#define NN 10000
#define EE 320000
#define HH 32
#define FIN 11
#define NEG_SLOPE 0.01f

typedef __attribute__((ext_vector_type(16))) __bf16 v16bf;
typedef __attribute__((ext_vector_type(8)))  float  v8f;

union ABfrag {
  v16bf  v;
  __bf16 e[16];
  uint4  q[2];
};

__device__ __forceinline__ v8f splat8(float x) {
  v8f r;
#pragma unroll
  for (int i = 0; i < 8; ++i) r[i] = x;
  return r;
}

__device__ __forceinline__ float relu_f(float x)  { return x > 0.0f ? x : 0.0f; }
__device__ __forceinline__ float leaky_f(float x) { return x > 0.0f ? x : NEG_SLOPE * x; }

// ---------------------------------------------------------------- zero ws
__global__ void zero_kernel(float* __restrict__ p, long n) {
  long i = (long)blockIdx.x * blockDim.x + threadIdx.x;
  long s = (long)gridDim.x * blockDim.x;
  for (; i < n; i += s) p[i] = 0.0f;
}

// ---------------------------------------------------------------- in-degree
__global__ void degree_kernel(const int* __restrict__ edges, float* __restrict__ cnt) {
  int i = blockIdx.x * blockDim.x + threadIdx.x;
  if (i < EE) atomicAdd(&cnt[edges[EE + i]], 1.0f);
}

// ---------------------------------------------------------------- node MLP: (B,N,11)f32 -> (B,N,32)bf16
__global__ void node_mlp_kernel(const float* __restrict__ V,
                                const float* __restrict__ W1, const float* __restrict__ b1,
                                const float* __restrict__ W2, const float* __restrict__ b2,
                                uint16_t* __restrict__ hb) {
  __shared__ float sW1[HH * FIN];
  __shared__ float sb1[HH];
  __shared__ float sW2[HH * HH];
  __shared__ float sb2[HH];
  for (int i = threadIdx.x; i < HH * FIN; i += blockDim.x) sW1[i] = W1[i];
  for (int i = threadIdx.x; i < HH * HH;  i += blockDim.x) sW2[i] = W2[i];
  if (threadIdx.x < HH) { sb1[threadIdx.x] = b1[threadIdx.x]; sb2[threadIdx.x] = b2[threadIdx.x]; }
  __syncthreads();

  int idx = blockIdx.x * blockDim.x + threadIdx.x;
  const int total = BB * NN;
  for (; idx < total; idx += gridDim.x * blockDim.x) {
    float x[FIN];
    const float* vp = V + (size_t)idx * FIN;
#pragma unroll
    for (int f = 0; f < FIN; ++f) x[f] = vp[f];
    float t[HH];
#pragma unroll 4
    for (int o = 0; o < HH; ++o) {
      float a = sb1[o];
#pragma unroll
      for (int f = 0; f < FIN; ++f) a += x[f] * sW1[o * FIN + f];
      t[o] = relu_f(a);
    }
    union { __bf16 e[HH]; uint4 q[4]; } outr;
#pragma unroll 4
    for (int o = 0; o < HH; ++o) {
      float a = sb2[o];
#pragma unroll
      for (int k = 0; k < HH; ++k) a += t[k] * sW2[o * HH + k];
      outr.e[o] = (__bf16)leaky_f(a);
    }
    uint4* dst = (uint4*)(hb + (size_t)idx * HH);
#pragma unroll
    for (int j = 0; j < 4; ++j) dst[j] = outr.q[j];
  }
}

// ---------------------------------------------------------------- edge MLP (WMMA) + scatter-add
// Per wave iteration: TWO tiles of 16 edges (32 edges). All gather loads issued
// before the 12-WMMA block so one s_wait_loadcnt covers the whole iteration.
//   GEMM1 (x2): D[16x32] = A[16x64 bf16]([h(src)|h(dst)]) x We1^T + be1 -> ReLU
//   GEMM2 (x2): D[16x32] = A[16x32 bf16](LDS relayout)    x We2^T + be2 -> LeakyReLU
//   scatter: global_atomic_add_f32 into agg[b, dst, :]
__global__ void __launch_bounds__(256)
edge_mlp_kernel(const uint16_t* __restrict__ hb, const int* __restrict__ edges,
                const float* __restrict__ We1, const float* __restrict__ be1,
                const float* __restrict__ We2, const float* __restrict__ be2,
                float* __restrict__ agg) {
  __shared__ __align__(16) uint16_t tileA2[8][2][16][32];  // per-wave two 16x32 bf16 tiles
  __shared__ __align__(16) int dsts[8][2][16];

  const int lane = threadIdx.x & 31;
  const int wave = threadIdx.x >> 5;
  const int m    = lane & 15;      // M-row (A/C) or N-col (B/C) owned by this lane
  const int hi   = lane >> 4;      // lane half
  const int mb   = hi * 8;         // C/D fragment: M = mb + r
  const int kA   = hi * 8;         // A fragment: K runs start at 8*hi and 16+8*hi
  const int kB   = hi * 16;        // unused; B fragments use the same K runs as A (via kA)

  // --- Weight B-fragments (built once per wave). B[K][N] = W[o=N][c=K].
  ABfrag bw1[2][2];  // [kstep][nhalf] for K=64 GEMM1
#pragma unroll
  for (int ks = 0; ks < 2; ++ks)
#pragma unroll
    for (int nh = 0; nh < 2; ++nh) {
      const float* wrow = We1 + (size_t)(nh * 16 + m) * 64 + ks * 32 + kA;
#pragma unroll
      for (int i = 0; i < 16; ++i) bw1[ks][nh].e[i] = (__bf16)wrow[i < 8 ? i : i + 8];
    }
  ABfrag bw2[2];     // [nhalf] for K=32 GEMM2
#pragma unroll
  for (int nh = 0; nh < 2; ++nh) {
    const float* wrow = We2 + (size_t)(nh * 16 + m) * 32 + kA;
#pragma unroll
    for (int i = 0; i < 16; ++i) bw2[nh].e[i] = (__bf16)wrow[i < 8 ? i : i + 8];
  }
  const float b1a = be1[m], b1b = be1[m + 16];
  const float b2a = be2[m], b2b = be2[m + 16];

  const int pairsPerB   = EE / 32;          // 10000 double-tiles per batch image
  const int totalPairs  = BB * pairsPerB;   // 80000
  const int nWaves      = gridDim.x * (blockDim.x >> 5);

  for (int p = blockIdx.x * (blockDim.x >> 5) + wave; p < totalPairs; p += nWaves) {
    const int b  = p / pairsPerB;
    const int e0 = (p - b * pairsPerB) * 32;

    // ---- gather indices for both tiles (lane handles row m of each)
    const int src0 = edges[e0 + m];
    const int dst0 = edges[EE + e0 + m];
    const int src1 = edges[e0 + 16 + m];
    const int dst1 = edges[EE + e0 + 16 + m];
    if (lane < 16) {
      dsts[wave][0][lane] = dst0;
      dsts[wave][1][lane] = dst1;
    }

    // ---- issue ALL A-fragment gathers (8 x global_load_b128) before any WMMA
    const uint16_t* base = hb + (size_t)b * NN * HH;
    const uint16_t* rs0 = base + (size_t)src0 * HH;
    const uint16_t* rd0 = base + (size_t)dst0 * HH;
    const uint16_t* rs1 = base + (size_t)src1 * HH;
    const uint16_t* rd1 = base + (size_t)dst1 * HH;
    ABfrag a0t0, a1t0, a0t1, a1t1;
    a0t0.q[0] = *(const uint4*)(rs0 + kA);
    a0t0.q[1] = *(const uint4*)(rs0 + 16 + kA);
    a1t0.q[0] = *(const uint4*)(rd0 + kA);
    a1t0.q[1] = *(const uint4*)(rd0 + 16 + kA);
    a0t1.q[0] = *(const uint4*)(rs1 + kA);
    a0t1.q[1] = *(const uint4*)(rs1 + 16 + kA);
    a1t1.q[0] = *(const uint4*)(rd1 + kA);
    a1t1.q[1] = *(const uint4*)(rd1 + 16 + kA);

    // ---- GEMM1 for both tiles: two independent 4-WMMA chains (ILP)
    v8f c0t0 = splat8(b1a), c1t0 = splat8(b1b);
    v8f c0t1 = splat8(b1a), c1t1 = splat8(b1b);
    c0t0 = __builtin_amdgcn_wmma_f32_16x16x32_bf16(false, a0t0.v, false, bw1[0][0].v, (short)0, c0t0, false, false);
    c0t1 = __builtin_amdgcn_wmma_f32_16x16x32_bf16(false, a0t1.v, false, bw1[0][0].v, (short)0, c0t1, false, false);
    c1t0 = __builtin_amdgcn_wmma_f32_16x16x32_bf16(false, a0t0.v, false, bw1[0][1].v, (short)0, c1t0, false, false);
    c1t1 = __builtin_amdgcn_wmma_f32_16x16x32_bf16(false, a0t1.v, false, bw1[0][1].v, (short)0, c1t1, false, false);
    c0t0 = __builtin_amdgcn_wmma_f32_16x16x32_bf16(false, a1t0.v, false, bw1[1][0].v, (short)0, c0t0, false, false);
    c0t1 = __builtin_amdgcn_wmma_f32_16x16x32_bf16(false, a1t1.v, false, bw1[1][0].v, (short)0, c0t1, false, false);
    c1t0 = __builtin_amdgcn_wmma_f32_16x16x32_bf16(false, a1t0.v, false, bw1[1][1].v, (short)0, c1t0, false, false);
    c1t1 = __builtin_amdgcn_wmma_f32_16x16x32_bf16(false, a1t1.v, false, bw1[1][1].v, (short)0, c1t1, false, false);

    // ---- ReLU + cross-lane relayout C(16x32 f32) -> A(16x32 bf16) via per-wave LDS
#pragma unroll
    for (int r = 0; r < 8; ++r) {
      tileA2[wave][0][mb + r][m]      = __builtin_bit_cast(unsigned short, (__bf16)relu_f(c0t0[r]));
      tileA2[wave][0][mb + r][m + 16] = __builtin_bit_cast(unsigned short, (__bf16)relu_f(c1t0[r]));
      tileA2[wave][1][mb + r][m]      = __builtin_bit_cast(unsigned short, (__bf16)relu_f(c0t1[r]));
      tileA2[wave][1][mb + r][m + 16] = __builtin_bit_cast(unsigned short, (__bf16)relu_f(c1t1[r]));
    }
    asm volatile("s_wait_dscnt 0" ::: "memory");  // LDS in-order per wave; compiler fence + wait

    ABfrag a2t0, a2t1;
    const uint16_t* row0 = &tileA2[wave][0][m][0];
    const uint16_t* row1 = &tileA2[wave][1][m][0];
    a2t0.q[0] = *(const uint4*)(row0 + kA);
    a2t0.q[1] = *(const uint4*)(row0 + 16 + kA);
    a2t1.q[0] = *(const uint4*)(row1 + kA);
    a2t1.q[1] = *(const uint4*)(row1 + 16 + kA);

    // ---- GEMM2 for both tiles
    v8f d0t0 = splat8(b2a), d1t0 = splat8(b2b);
    v8f d0t1 = splat8(b2a), d1t1 = splat8(b2b);
    d0t0 = __builtin_amdgcn_wmma_f32_16x16x32_bf16(false, a2t0.v, false, bw2[0].v, (short)0, d0t0, false, false);
    d0t1 = __builtin_amdgcn_wmma_f32_16x16x32_bf16(false, a2t1.v, false, bw2[0].v, (short)0, d0t1, false, false);
    d1t0 = __builtin_amdgcn_wmma_f32_16x16x32_bf16(false, a2t0.v, false, bw2[1].v, (short)0, d1t0, false, false);
    d1t1 = __builtin_amdgcn_wmma_f32_16x16x32_bf16(false, a2t1.v, false, bw2[1].v, (short)0, d1t1, false, false);

    // ---- destination ids for this lane's 8 rows: two b128 LDS loads per tile
    union { uint4 q; int i[4]; } dq0a, dq0b, dq1a, dq1b;
    dq0a.q = *(const uint4*)(&dsts[wave][0][mb]);
    dq0b.q = *(const uint4*)(&dsts[wave][0][mb + 4]);
    dq1a.q = *(const uint4*)(&dsts[wave][1][mb]);
    dq1b.q = *(const uint4*)(&dsts[wave][1][mb + 4]);

    // ---- LeakyReLU + scatter-add: lane owns columns N=m and m+16, rows M=mb+r
    float* aggb = agg + (size_t)b * NN * HH;
#pragma unroll
    for (int r = 0; r < 8; ++r) {
      const int dn0 = (r < 4) ? dq0a.i[r] : dq0b.i[r - 4];
      float* outp0 = aggb + (size_t)dn0 * HH;
      atomicAdd(outp0 + m,      leaky_f(d0t0[r]));
      atomicAdd(outp0 + m + 16, leaky_f(d1t0[r]));
      const int dn1 = (r < 4) ? dq1a.i[r] : dq1b.i[r - 4];
      float* outp1 = aggb + (size_t)dn1 * HH;
      atomicAdd(outp1 + m,      leaky_f(d0t1[r]));
      atomicAdd(outp1 + m + 16, leaky_f(d1t1[r]));
    }
    asm volatile("" ::: "memory");
  }
}

// ---------------------------------------------------------------- vertex MLP + graph readout partials
__global__ void vertex_readout_kernel(const float* __restrict__ agg, const float* __restrict__ cnt,
                                      const float* __restrict__ Wv1, const float* __restrict__ bv1,
                                      const float* __restrict__ Wv2, const float* __restrict__ bv2,
                                      const float* __restrict__ Wg, float* __restrict__ acc) {
  __shared__ float sW1[HH * HH];
  __shared__ float sb1[HH];
  __shared__ float sW2[HH];
  __shared__ float sbias2;
  __shared__ float sacc[BB];
  for (int i = threadIdx.x; i < HH * HH; i += blockDim.x) sW1[i] = Wv1[i];
  if (threadIdx.x < HH) { sb1[threadIdx.x] = bv1[threadIdx.x]; sW2[threadIdx.x] = Wv2[threadIdx.x]; }
  if (threadIdx.x == 0) sbias2 = bv2[0];
  if (threadIdx.x < BB) sacc[threadIdx.x] = 0.0f;
  __syncthreads();

  const int idx = blockIdx.x * blockDim.x + threadIdx.x;
  if (idx < BB * NN) {
    const int b = idx / NN;
    const int n = idx - b * NN;
    const float inv = 1.0f / fmaxf(cnt[n], 1.0f);
    float a[HH];
    const float* ap = agg + (size_t)idx * HH;
#pragma unroll
    for (int k = 0; k < HH; ++k) a[k] = ap[k] * inv;
    float v = sbias2;
#pragma unroll 4
    for (int o = 0; o < HH; ++o) {
      float t = sb1[o];
#pragma unroll
      for (int k = 0; k < HH; ++k) t += a[k] * sW1[o * HH + k];
      v += relu_f(t) * sW2[o];
    }
    v = leaky_f(v);
    atomicAdd(&sacc[b], v * Wg[n]);
  }
  __syncthreads();
  if (threadIdx.x < BB) atomicAdd(&acc[threadIdx.x], sacc[threadIdx.x]);
}

// ---------------------------------------------------------------- final sigmoid
__global__ void sigmoid_kernel(const float* __restrict__ acc, const float* __restrict__ bg,
                               float* __restrict__ out) {
  const int t = threadIdx.x;
  if (t < BB) out[t] = 1.0f / (1.0f + expf(-(acc[t] + bg[0])));
}

// ---------------------------------------------------------------- launch
extern "C" void kernel_launch(void* const* d_in, const int* in_sizes, int n_in,
                              void* d_out, int out_size, void* d_ws, size_t ws_size,
                              hipStream_t stream) {
  const float* verts = (const float*)d_in[0];
  const int*   edges = (const int*)d_in[1];
  const float* Wx1 = (const float*)d_in[2];  const float* bx1 = (const float*)d_in[3];
  const float* Wx2 = (const float*)d_in[4];  const float* bx2 = (const float*)d_in[5];
  const float* We1 = (const float*)d_in[6];  const float* be1 = (const float*)d_in[7];
  const float* We2 = (const float*)d_in[8];  const float* be2 = (const float*)d_in[9];
  const float* Wv1 = (const float*)d_in[10]; const float* bv1 = (const float*)d_in[11];
  const float* Wv2 = (const float*)d_in[12]; const float* bv2 = (const float*)d_in[13];
  const float* Wg  = (const float*)d_in[14]; const float* bg  = (const float*)d_in[15];

  // Workspace layout (16B aligned throughout):
  //   agg: B*N*32 f32 | cnt: N f32 | acc: B f32 | h: B*N*32 bf16
  float*    agg = (float*)d_ws;
  float*    cnt = agg + (size_t)BB * NN * HH;
  float*    acc = cnt + NN;
  uint16_t* hb  = (uint16_t*)(acc + BB);
  float*    out = (float*)d_out;

  const long zn = (long)BB * NN * HH + NN + BB;
  zero_kernel<<<2048, 256, 0, stream>>>(agg, zn);
  node_mlp_kernel<<<(BB * NN + 255) / 256, 256, 0, stream>>>(verts, Wx1, bx1, Wx2, bx2, hb);
  degree_kernel<<<(EE + 255) / 256, 256, 0, stream>>>(edges, cnt);
  edge_mlp_kernel<<<1920, 256, 0, stream>>>(hb, edges, We1, be1, We2, be2, agg);
  vertex_readout_kernel<<<(BB * NN + 255) / 256, 256, 0, stream>>>(agg, cnt, Wv1, bv1, Wv2, bv2, Wg, acc);
  sigmoid_kernel<<<1, BB, 0, stream>>>(acc, bg, out);

  (void)in_sizes; (void)n_in; (void)out_size; (void)ws_size;
}
